// AmplitudeWeightedPhaseAttention_77618648973384
// MI455X (gfx1250) — matrix operand order repaired and checked by reading, not compiled, not run on hardware
//
#include <hip/hip_runtime.h>
#include <math.h>

// ---------------------------------------------------------------------------
// AmplitudeWeightedPhaseAttention for MI455X (gfx1250, wave32, WMMA).
//
// Factorization:  cos(dphi)*sqrt(|Q||K|) = xq*xk + yq*yk  with
//   x = Re/sqrt(|.|), y = Im/sqrt(|.|), a = sqrt(|.|)
// so the O(S^2 * F) pairwise stage becomes two f16 WMMA GEMMs (K=96 / K=160
// after zero padding), and the output stage is a third WMMA GEMM.
// ---------------------------------------------------------------------------

typedef _Float16 v16h __attribute__((ext_vector_type(16)));
typedef _Float16 v8h  __attribute__((ext_vector_type(8)));
typedef float    v8f  __attribute__((ext_vector_type(8)));

#define B_N 2
#define S_N 1024
#define H_N 128
#define F_N 65
#define KA_N 96     // padded K for amplitude GEMM  (65 -> 96, 3 wmma steps)
#define KP_N 160    // padded K for phase GEMM      (130 -> 160, 5 wmma steps)

__device__ __forceinline__ v16h ld_frag(const _Float16* p0, const _Float16* p1) {
  v8h lo = *(const v8h*)p0;
  v8h hi = *(const v8h*)p1;
  return __builtin_shufflevector(lo, hi, 0,1,2,3,4,5,6,7,8,9,10,11,12,13,14,15);
}

// ---------------------------------------------------------------------------
// 1) per-row 128-point real DFT of Q and K; emit f16 WMMA operand arrays.
//    blockIdx.x = b*S + s, blockIdx.y = 0 (Q) / 1 (K), 128 threads.
// ---------------------------------------------------------------------------
__global__ void prep_fft(const float* __restrict__ Q, const float* __restrict__ K,
                         _Float16* __restrict__ QA, _Float16* __restrict__ KA,
                         _Float16* __restrict__ QP, _Float16* __restrict__ KP) {
  const int row = blockIdx.x;
  const int isK = blockIdx.y;
  const float* src = (isK ? K : Q) + (size_t)row * H_N;
  _Float16* Aout = (isK ? KA : QA) + (size_t)row * KA_N;
  _Float16* Pout = (isK ? KP : QP) + (size_t)row * KP_N;

  __shared__ float xs[H_N];
  const int t = threadIdx.x;            // 0..127
  xs[t] = src[t];
  __syncthreads();

  if (t < F_N) {
    const float w0 = (-2.0f * 3.14159265358979323846f / (float)H_N) * (float)t;
    float re = 0.f, im = 0.f;
    for (int h = 0; h < H_N; ++h) {
      float sa, ca;
      __sincosf(w0 * (float)h, &sa, &ca);
      re = fmaf(xs[h], ca, re);
      im = fmaf(xs[h], sa, im);
    }
    const float amp = sqrtf(re * re + im * im + 1e-20f);
    const float rs  = rsqrtf(amp);        // 1/sqrt(amp)
    Aout[t]       = (_Float16)(amp * rs); // sqrt(amp)
    Pout[t]       = (_Float16)(re * rs);  // amp^.5 * cos(phase)
    Pout[F_N + t] = (_Float16)(im * rs);  // amp^.5 * sin(phase)
  } else if (t < KA_N) {
    Aout[t] = (_Float16)0.f;                         // pad 65..95
  } else if (t < KA_N + (KP_N - 2 * F_N)) {
    Pout[2 * F_N + (t - KA_N)] = (_Float16)0.f;      // pad 130..159
  }
}

// ---------------------------------------------------------------------------
// 2) V -> f16, transposed to [B, H, S] so GEMM B-fragments are K-contiguous.
// ---------------------------------------------------------------------------
__global__ void prep_v(const float* __restrict__ V, _Float16* __restrict__ VT) {
  const int row = blockIdx.x;           // b*S + s
  const int b = row / S_N, s = row % S_N;
  const int h = threadIdx.x;            // 0..127
  VT[((size_t)b * H_N + h) * S_N + s] = (_Float16)V[(size_t)row * H_N + h];
}

// ---------------------------------------------------------------------------
// 3) scores GEMM: one 16x16 tile per wave, 8 waves per block.
//    c1 = sum_f aq*ak (3 wmma), c2 = sum xq*xk + yq*yk (5 wmma),
//    score = c2/(c1+1e-8) + 1, stored into the weights region of d_out.
// ---------------------------------------------------------------------------
__global__ void scores_gemm(const _Float16* __restrict__ QA, const _Float16* __restrict__ KA,
                            const _Float16* __restrict__ QP, const _Float16* __restrict__ KP,
                            float* __restrict__ SC) {
  const int lane = threadIdx.x & 31;
  const int wave = threadIdx.x >> 5;
  const int tile = blockIdx.x * 8 + wave;   // 0..8191
  const int b  = tile >> 12;                // tile / 4096
  const int ti = (tile >> 6) & 63;
  const int tj = tile & 63;
  const int i0 = ti * 16, j0 = tj * 16;
  const int lh = lane & 15;
  const int half = lane >> 4;

  // amplitude GEMM (K = 96)
  const _Float16* aA = QA + (size_t)(b * S_N + i0 + lh) * KA_N + half * 16;
  const _Float16* bA = KA + (size_t)(b * S_N + j0 + lh) * KA_N + half * 16;
  v8f c1 = {};
  #pragma unroll
  for (int kc = 0; kc < KA_N; kc += 32) {
    v16h a  = ld_frag(aA + kc, aA + kc + 8);
    v16h bb = ld_frag(bA + kc, bA + kc + 8);
    c1 = __builtin_amdgcn_wmma_f32_16x16x32_f16(false, a, false, bb, (short)0, c1, false, false);
  }
  // phase GEMM (K = 160)
  const _Float16* aP = QP + (size_t)(b * S_N + i0 + lh) * KP_N + half * 16;
  const _Float16* bP = KP + (size_t)(b * S_N + j0 + lh) * KP_N + half * 16;
  v8f c2 = {};
  #pragma unroll
  for (int kc = 0; kc < KP_N; kc += 32) {
    v16h a  = ld_frag(aP + kc, aP + kc + 8);
    v16h bb = ld_frag(bP + kc, bP + kc + 8);
    c2 = __builtin_amdgcn_wmma_f32_16x16x32_f16(false, a, false, bb, (short)0, c2, false, false);
  }

  float* outCol = SC + (size_t)b * S_N * S_N + (size_t)(j0 + lh);
  #pragma unroll
  for (int r = 0; r < 8; ++r) {
    const int m = i0 + r + (half ? 8 : 0);
    outCol[(size_t)m * S_N] = c2[r] / (c1[r] + 1e-8f) + 1.0f;
  }
}

// ---------------------------------------------------------------------------
// 4) in-place row softmax over 1024 scores; also emit f16 copy for final GEMM.
// ---------------------------------------------------------------------------
__global__ void softmax_k(float* __restrict__ SC, _Float16* __restrict__ WF) {
  const int row = blockIdx.x;                     // b*S + i
  float*    p  = SC + (size_t)row * S_N;
  _Float16* pf = WF + (size_t)row * S_N;
  const int t = threadIdx.x;                      // 0..255
  __shared__ float red[256];

  float v0 = p[t], v1 = p[t + 256], v2 = p[t + 512], v3 = p[t + 768];
  float m = fmaxf(fmaxf(v0, v1), fmaxf(v2, v3));
  red[t] = m; __syncthreads();
  for (int s = 128; s > 0; s >>= 1) {
    if (t < s) red[t] = fmaxf(red[t], red[t + s]);
    __syncthreads();
  }
  m = red[0]; __syncthreads();

  float e0 = __expf(v0 - m), e1 = __expf(v1 - m), e2 = __expf(v2 - m), e3 = __expf(v3 - m);
  red[t] = e0 + e1 + e2 + e3; __syncthreads();
  for (int s = 128; s > 0; s >>= 1) {
    if (t < s) red[t] += red[t + s];
    __syncthreads();
  }
  const float inv = 1.0f / red[0];

  const float w0 = e0 * inv, w1 = e1 * inv, w2 = e2 * inv, w3 = e3 * inv;
  p[t] = w0;        p[t + 256] = w1;        p[t + 512] = w2;        p[t + 768] = w3;
  pf[t] = (_Float16)w0; pf[t + 256] = (_Float16)w1;
  pf[t + 512] = (_Float16)w2; pf[t + 768] = (_Float16)w3;
}

// ---------------------------------------------------------------------------
// 5) output GEMM: out = weights(f16) @ V  ([S,S] x [S,H]), 32 wmma k-steps.
// ---------------------------------------------------------------------------
__global__ void out_gemm(const _Float16* __restrict__ WF, const _Float16* __restrict__ VT,
                         float* __restrict__ OUT) {
  const int lane = threadIdx.x & 31;
  const int wave = threadIdx.x >> 5;
  const int tile = blockIdx.x * 8 + wave;   // 0..1023
  const int b  = tile >> 9;                 // / (64*8)
  const int ti = (tile >> 3) & 63;
  const int th = tile & 7;
  const int i0 = ti * 16, h0 = th * 16;
  const int lh = lane & 15, half = lane >> 4;

  const _Float16* aBase = WF + (size_t)b * S_N * S_N + (size_t)(i0 + lh) * S_N + half * 16;
  const _Float16* bBase = VT + (size_t)b * H_N * S_N + (size_t)(h0 + lh) * S_N + half * 16;
  v8f c = {};
  for (int kc = 0; kc < S_N; kc += 32) {
    v16h a  = ld_frag(aBase + kc, aBase + kc + 8);
    v16h bb = ld_frag(bBase + kc, bBase + kc + 8);
    c = __builtin_amdgcn_wmma_f32_16x16x32_f16(false, a, false, bb, (short)0, c, false, false);
  }
  float* o = OUT + (size_t)(b * S_N + i0) * H_N + (h0 + lh);
  #pragma unroll
  for (int r = 0; r < 8; ++r) {
    o[(size_t)(r + (half ? 8 : 0)) * H_N] = c[r];
  }
}

// ---------------------------------------------------------------------------
extern "C" void kernel_launch(void* const* d_in, const int* in_sizes, int n_in,
                              void* d_out, int out_size, void* d_ws, size_t ws_size,
                              hipStream_t stream) {
  const float* Q = (const float*)d_in[0];
  const float* K = (const float*)d_in[1];
  const float* V = (const float*)d_in[2];

  float* out = (float*)d_out;                       // [B,S,H] f32
  float* wts = out + (size_t)B_N * S_N * H_N;       // [B,S,S] f32 (output #2)

  char* ws = (char*)d_ws;                           // ~6.5 MB total
  _Float16* QA = (_Float16*)(ws + 0);               //  393216 B
  _Float16* KA = (_Float16*)(ws + 393216);          //  393216 B
  _Float16* QP = (_Float16*)(ws + 786432);          //  655360 B
  _Float16* KP = (_Float16*)(ws + 1441792);         //  655360 B
  _Float16* VT = (_Float16*)(ws + 2097152);         //  524288 B
  _Float16* WF = (_Float16*)(ws + 2621440);         // 4194304 B

  prep_fft  <<<dim3(B_N * S_N, 2), 128, 0, stream>>>(Q, K, QA, KA, QP, KP);
  prep_v    <<<B_N * S_N, H_N, 0, stream>>>(V, VT);
  scores_gemm<<<1024, 256, 0, stream>>>(QA, KA, QP, KP, wts);
  softmax_k <<<B_N * S_N, 256, 0, stream>>>(wts, WF);
  out_gemm  <<<128, 256, 0, stream>>>(WF, VT, out);
}
